// Crosscoder_73967926771810
// MI455X (gfx1250) — compile-verified, matrix operand order checked by reading, not run on hardware
//
#include <hip/hip_runtime.h>

// ---------------------------------------------------------------------------
// Crosscoder: encode GEMM -> top-k(64)/relu -> two decode GEMMs.
// B=4096, D_IN=2048, D_HID=16384, K=64.
// bf16 WMMA GEMM with ASYNC global->LDS staging (double-buffered fp32 tiles,
// K-tile = 64 so each wait+barrier covers 16 WMMAs/wave), native bf16
// converts at fragment build. pre written into d_out's z region, masked in
// place by an exact radix top-k selector.
// ---------------------------------------------------------------------------

typedef unsigned short u16;
typedef __attribute__((ext_vector_type(16))) __bf16 v16bf;
typedef __attribute__((ext_vector_type(8)))  float  v8f;

#define TILE_M 128
#define TILE_N 128
#define TILE_K 64
#define ASTR   72      // fp32 elements per LDS tile row (64 + 8 pad, keeps 16B align)

#define BATCH 4096
#define D_IN  2048
#define D_HID 16384
#define KSEL  64

__device__ __forceinline__ void wait_async0() {
#if __has_builtin(__builtin_amdgcn_s_wait_asynccnt)
  __builtin_amdgcn_s_wait_asynccnt(0);
#else
  asm volatile("s_wait_asynccnt 0x0" ::: "memory");
#endif
}

// Copy 128 contiguous bytes global -> LDS asynchronously (8 x b128 granules).
// The instruction offset is added to both the LDS and the global address, so one address
// setup serves all eight granules via the instruction offset field.
__device__ __forceinline__ void async_row128(unsigned ldsOff, const float* gp) {
  asm volatile(
      "global_load_async_to_lds_b128 %0, %1, off\n\t"
      "global_load_async_to_lds_b128 %0, %1, off offset:16\n\t"
      "global_load_async_to_lds_b128 %0, %1, off offset:32\n\t"
      "global_load_async_to_lds_b128 %0, %1, off offset:48\n\t"
      "global_load_async_to_lds_b128 %0, %1, off offset:64\n\t"
      "global_load_async_to_lds_b128 %0, %1, off offset:80\n\t"
      "global_load_async_to_lds_b128 %0, %1, off offset:96\n\t"
      "global_load_async_to_lds_b128 %0, %1, off offset:112"
      :: "v"(ldsOff), "v"((unsigned long long)(size_t)gp)
      : "memory");
}

__device__ __forceinline__ unsigned f2key(float f) {
  // monotonic float -> uint mapping (order-preserving, incl. negatives)
  unsigned b = __float_as_uint(f);
  unsigned m = (unsigned)((int)b >> 31) | 0x80000000u;
  return b ^ m;
}

// ---------------------------------------------------------------------------
// Generic 128x128 bf16-WMMA GEMM:  C[m][n] = sum_k A(m,k) * Bmat[n][k] (+bias)
// A(m,k) = (k < kSplit) ? A0[m*lda + k] : A1[m*lda + (k - kSplit)]
// grid = (M/128, N/128), 256 threads = 8 waves; wave w owns a 2(M)x4(N) grid
// of 16x16 accumulators. Double-buffered async LDS staging, 64-wide K tiles.
// ---------------------------------------------------------------------------
__global__ __launch_bounds__(256)
void gemm_bf16_wmma(const float* __restrict__ A0, const float* __restrict__ A1,
                    int kSplit, int lda,
                    const float* __restrict__ Bmat, int ldb,
                    const float* __restrict__ bias,
                    float* __restrict__ C, int ldc, int Kdim)
{
  __shared__ __align__(16) float bufA[2][TILE_M * ASTR];
  __shared__ __align__(16) float bufB[2][TILE_N * ASTR];

  const int t     = threadIdx.x;
  const int lane  = t & 31;
  const int wave  = t >> 5;
  const int mBase = blockIdx.x * TILE_M;
  const int nBase = blockIdx.y * TILE_N;

  // staging: thread t owns row sr, 32-float half sh (128 bytes = 8 granules)
  const int sr = t >> 1;
  const int sh = (t & 1) << 5;

  // wave tiling
  const int wm  = wave & 3;      // M band: rows wm*32 .. +32
  const int wn  = wave >> 2;     // N band: cols wn*64 .. +64
  const int fr  = lane & 15;     // row within 16x16 fragment
  const int kg  = lane >> 4;     // K-group select (ISA 16-bit A/B layout)
  const int klo = kg * 8;
  const int khi = 16 + kg * 8;

  const float* aRow  = A0 + (size_t)(mBase + sr) * lda;   // same lda both halves
  const float* aRow1 = A1 + (size_t)(mBase + sr) * lda;
  const float* bRow  = Bmat + (size_t)(nBase + sr) * ldb;

  auto stage = [&](int k0, int cur) {
    int gk = k0 + sh;            // kSplit is a multiple of 32, never straddled
    const float* gpA = (gk < kSplit) ? (aRow + gk) : (aRow1 + (gk - kSplit));
    async_row128((unsigned)(size_t)&bufA[cur][sr * ASTR + sh], gpA);
    async_row128((unsigned)(size_t)&bufB[cur][sr * ASTR + sh], bRow + (k0 + sh));
  };

  v8f zeroacc = {};
  v8f acc[2][4];
#pragma unroll
  for (int i = 0; i < 2; ++i)
#pragma unroll
    for (int j = 0; j < 4; ++j)
      acc[i][j] = zeroacc;

  stage(0, 0);   // prologue: fill buffer 0

  for (int k0 = 0; k0 < Kdim; k0 += TILE_K) {
    const int cur = (k0 >> 6) & 1;

    wait_async0();       // my async copies into bufs[cur] are done
    __syncthreads();     // everyone's copies are done; prev reads retired

    if (k0 + TILE_K < Kdim) stage(k0 + TILE_K, cur ^ 1);   // overlap next tile

#pragma unroll
    for (int half = 0; half < 2; ++half) {
      const int kb = half * 32;

      // ---- build fragments (fp32 LDS -> native bf16 converts) ----
      v16bf aF[2];
#pragma unroll
      for (int i = 0; i < 2; ++i) {
        const float* ap = &bufA[cur][(wm * 32 + i * 16 + fr) * ASTR + kb];
        float4 q0 = *(const float4*)(ap + klo);
        float4 q1 = *(const float4*)(ap + klo + 4);
        float4 q2 = *(const float4*)(ap + khi);
        float4 q3 = *(const float4*)(ap + khi + 4);
        v16bf r = { (__bf16)q0.x, (__bf16)q0.y, (__bf16)q0.z, (__bf16)q0.w,
                    (__bf16)q1.x, (__bf16)q1.y, (__bf16)q1.z, (__bf16)q1.w,
                    (__bf16)q2.x, (__bf16)q2.y, (__bf16)q2.z, (__bf16)q2.w,
                    (__bf16)q3.x, (__bf16)q3.y, (__bf16)q3.z, (__bf16)q3.w };
        aF[i] = r;
      }
      v16bf bF[4];
#pragma unroll
      for (int j = 0; j < 4; ++j) {
        const float* bp = &bufB[cur][(wn * 64 + j * 16 + fr) * ASTR + kb];
        float4 q0 = *(const float4*)(bp + klo);
        float4 q1 = *(const float4*)(bp + klo + 4);
        float4 q2 = *(const float4*)(bp + khi);
        float4 q3 = *(const float4*)(bp + khi + 4);
        v16bf r = { (__bf16)q0.x, (__bf16)q0.y, (__bf16)q0.z, (__bf16)q0.w,
                    (__bf16)q1.x, (__bf16)q1.y, (__bf16)q1.z, (__bf16)q1.w,
                    (__bf16)q2.x, (__bf16)q2.y, (__bf16)q2.z, (__bf16)q2.w,
                    (__bf16)q3.x, (__bf16)q3.y, (__bf16)q3.z, (__bf16)q3.w };
        bF[j] = r;
      }

      // ---- 8 WMMAs per half: D = A*B + C, f32 accumulate ----
#pragma unroll
      for (int i = 0; i < 2; ++i)
#pragma unroll
        for (int j = 0; j < 4; ++j)
          acc[i][j] = __builtin_amdgcn_wmma_f32_16x16x32_bf16(
              false, aF[i], false, bF[j], (short)0, acc[i][j], false, false);
    }
  }

  // ---- epilogue: C-fragment layout (lanes 0-15: M=r, lanes 16-31: M=r+8) ----
#pragma unroll
  for (int i = 0; i < 2; ++i) {
#pragma unroll
    for (int j = 0; j < 4; ++j) {
      int col = nBase + wn * 64 + j * 16 + fr;
      float bv = bias ? bias[col] : 0.0f;
      int rowTop = mBase + wm * 32 + i * 16 + kg * 8;
#pragma unroll
      for (int r = 0; r < 8; ++r)
        C[(size_t)(rowTop + r) * ldc + col] = acc[i][j][r] + bv;
    }
  }
}

// ---------------------------------------------------------------------------
// Exact top-64-of-16384 per row via 8-bit radix select on monotonic keys,
// then in-place masked ReLU write. One 256-thread block per row.
// ---------------------------------------------------------------------------
__global__ __launch_bounds__(256)
void topk_relu_kernel(float* __restrict__ z)
{
  const int row = blockIdx.x;
  float* p = z + (size_t)row * D_HID;

  __shared__ int hist[256];
  __shared__ unsigned sPrefix;
  __shared__ int sKRem;
  __shared__ int sTie;

  const int t = threadIdx.x;
  if (t == 0) { sPrefix = 0u; sKRem = KSEL; sTie = 0; }
  __syncthreads();

  for (int shift = 24; shift >= 0; shift -= 8) {
    hist[t] = 0;
    __syncthreads();
    unsigned prefix = sPrefix;
    unsigned hmask  = (shift == 24) ? 0u : (0xFFFFFFFFu << (shift + 8));
    for (int i = t; i < D_HID; i += 256) {
      unsigned kk = f2key(p[i]);
      if ((kk & hmask) == (prefix & hmask))
        atomicAdd(&hist[(kk >> shift) & 255], 1);
    }
    __syncthreads();
    if (t == 0) {
      int kRem = sKRem, cum = 0, b;
      for (b = 255; b > 0; --b) {
        if (cum + hist[b] >= kRem) break;
        cum += hist[b];
      }
      sPrefix = prefix | ((unsigned)b << shift);
      sKRem   = kRem - cum;
    }
    __syncthreads();
  }

  const unsigned T = sPrefix;   // key of the 64th-largest element
  const int budget = sKRem;     // ties at T to include
  for (int i = t; i < D_HID; i += 256) {
    float v = p[i];
    unsigned kk = f2key(v);
    float outv = 0.0f;
    if (kk > T) {
      outv = fmaxf(v, 0.0f);
    } else if (kk == T) {
      if (atomicAdd(&sTie, 1) < budget) outv = fmaxf(v, 0.0f);
    }
    p[i] = outv;
  }
}

// ---------------------------------------------------------------------------
extern "C" void kernel_launch(void* const* d_in, const int* in_sizes, int n_in,
                              void* d_out, int out_size, void* d_ws, size_t ws_size,
                              hipStream_t stream) {
  const float* x_m     = (const float*)d_in[0];   // [4096, 2048]
  const float* x_p     = (const float*)d_in[1];   // [4096, 2048]
  const float* W_enc   = (const float*)d_in[2];   // [16384, 4096]
  const float* b_enc   = (const float*)d_in[3];   // [16384]
  const float* W_dec_m = (const float*)d_in[4];   // [2048, 16384]
  const float* W_dec_p = (const float*)d_in[5];   // [2048, 16384]
  // d_in[6] = k (device scalar); K=64 is fixed by the reference.

  float* rec_m = (float*)d_out;                           // [4096, 2048]
  float* rec_p = rec_m + (size_t)BATCH * D_IN;            // [4096, 2048]
  float* z     = rec_p + (size_t)BATCH * D_IN;            // [4096, 16384]

  // 1) encode: pre = [x_m | x_p] @ W_enc^T + b_enc  -> written into z region
  gemm_bf16_wmma<<<dim3(BATCH / TILE_M, D_HID / TILE_N), 256, 0, stream>>>(
      x_m, x_p, /*kSplit=*/D_IN, /*lda=*/D_IN,
      W_enc, /*ldb=*/2 * D_IN, b_enc,
      z, /*ldc=*/D_HID, /*Kdim=*/2 * D_IN);

  // 2) exact top-64 + relu, in place on z
  topk_relu_kernel<<<dim3(BATCH), 256, 0, stream>>>(z);

  // 3) decode twice: rec = z @ W_dec^T
  gemm_bf16_wmma<<<dim3(BATCH / TILE_M, D_IN / TILE_N), 256, 0, stream>>>(
      z, z, /*kSplit=*/D_HID, /*lda=*/D_HID,
      W_dec_m, /*ldb=*/D_HID, nullptr,
      rec_m, /*ldc=*/D_IN, /*Kdim=*/D_HID);
  gemm_bf16_wmma<<<dim3(BATCH / TILE_M, D_IN / TILE_N), 256, 0, stream>>>(
      z, z, /*kSplit=*/D_HID, /*lda=*/D_HID,
      W_dec_p, /*ldb=*/D_HID, nullptr,
      rec_p, /*ldc=*/D_IN, /*Kdim=*/D_HID);
}
